// ResidualIntegrationNetworkRK4_19619410608506
// MI455X (gfx1250) — matrix operand order repaired and checked by reading, not compiled, not run on hardware
//
#include <hip/hip_runtime.h>

#define S_LEN 8192
#define B_SZ  1024
#define F_CT  2
#define H_W   32

typedef float    v8f  __attribute__((ext_vector_type(8)));
typedef float    v2f  __attribute__((ext_vector_type(2)));
typedef unsigned v8u  __attribute__((ext_vector_type(8)));
typedef _Float16 v16h __attribute__((ext_vector_type(16)));
typedef __fp16   v2h  __attribute__((ext_vector_type(2)));   // cvt_pkrtz return type

// gfx1250 hardware tanh (transcendental VALU op).
static __device__ __forceinline__ float tanh_hw(float x) {
#if __has_builtin(__builtin_amdgcn_tanhf)
    return __builtin_amdgcn_tanhf(x);
#else
    float r;
    asm("v_tanh_f32 %0, %1" : "=v"(r) : "v"(x));
    return r;
#endif
}

// pack two f32 -> packed f16 dword (v_cvt_pk_rtz_f16_f32)
static __device__ __forceinline__ unsigned pk16(float a, float b) {
    v2h p = __builtin_amdgcn_cvt_pkrtz(a, b);
    return __builtin_bit_cast(unsigned, p);
}

// Exchange value with lane^16 partner (pure-VALU cross-half swap).
static __device__ __forceinline__ unsigned xhalf_u(unsigned v) {
#if __has_builtin(__builtin_amdgcn_permlanex16)
    return __builtin_amdgcn_permlanex16(v, v, 0x76543210u, 0xFEDCBA98u, false, false);
#else
    return (unsigned)__shfl_xor((int)v, 16, 32);
#endif
}
static __device__ __forceinline__ float xhalf_f(float v) {
    return __builtin_bit_cast(float, xhalf_u(__builtin_bit_cast(unsigned, v)));
}

// One wave (32 lanes) integrates 16 batch columns for all S steps.
// All matrices kept transposed (hidden x batch).
//
// Hidden-layer-1 units are RELABELED so that stage-1's f32 WMMA D-layout,
// after tanh + f16 packing, is bit-for-bit the stage-2 f16 WMMA B-layout:
//   B position K in this code's k order, A matches (lane<16: K=0..15, lane>=16: K=16..31; two K per VGPR)
//   is produced at stage-1 source: half-a rows carry K in {0..7}u{16..23},
//   half-b rows carry K in {8..15}u{24..31}. W1/b1/w1y are permuted to match;
//   W2's K-gather already indexes by B position. No cross-lane fix-up needed.
__global__ __launch_bounds__(32)
void rk4_ode_wmma_kernel(const float* __restrict__ x,
                         const float* __restrict__ W1, const float* __restrict__ b1,
                         const float* __restrict__ W2, const float* __restrict__ b2,
                         const float* __restrict__ W3, const float* __restrict__ b3,
                         float* __restrict__ out)
{
    const int  lane = threadIdx.x & 31;
    const bool lo   = lane < 16;
    const int  M    = lane & 15;          // A-matrix row / batch column index
    const int  m8   = lo ? 0 : 8;         // D/C register->row offset for hi lanes
    const int  b0   = blockIdx.x * 16;    // batch tile base

    // Permuted hid-1 unit for stage-1 A row M:  half-a: M<8?M:M+8, half-b: +8
    const int uA = (M < 8) ? M : (M + 8);
    const int uB = uA + 8;

    // ---- Stage-1 A matrices: W1^T x-feature columns, f32 16x4 layout ----
    // lane<16 holds K=0,1 (features); lane>=16 holds K=2,3 (zero-padded:
    // the y contribution is applied as a rank-1 VALU update instead).
    v2f a1a, a1b;
    a1a[0] = lo ? W1[0 * H_W + uA] : 0.f;
    a1a[1] = lo ? W1[1 * H_W + uA] : 0.f;
    a1b[0] = lo ? W1[0 * H_W + uB] : 0.f;
    a1b[1] = lo ? W1[1 * H_W + uB] : 0.f;

    // ---- Stage-2 A matrices: W2^T halves, f16 16x32 A layout ----
    // K here is the B-matrix position == permuted hid-1 unit index, which is
    // exactly the identity gather on W2's rows (units == positions).
    v16h a2a, a2b;
    {
        const int kb = lo ? 0 : 16;
        #pragma unroll
        for (int i = 0; i < 16; ++i) {
            const int K = kb + i;
            a2a[i] = (_Float16)W2[K * H_W + M];
            a2b[i] = (_Float16)W2[K * H_W + 16 + M];
        }
    }

    // ---- Biases / rank-1 vectors in (permuted) D-layout row order ----
    // Stage-1 D half-a reg r: unit = lo ? r : r+16 ; half-b: +8.
    v8f cb1a, cb1b, cb2a, cb2b;
    float w1y[16], w3r[16];
    #pragma unroll
    for (int r = 0; r < 8; ++r) {
        const int hA = lo ? r : (r + 16);
        cb1a[r]    = b1[hA];
        cb1b[r]    = b1[hA + 8];
        w1y[r]     = W1[2 * H_W + hA];
        w1y[8 + r] = W1[2 * H_W + hA + 8];
        // hid-out ordering is untouched (plain D-layout map)
        cb2a[r]    = b2[r + m8];
        cb2b[r]    = b2[16 + r + m8];
        w3r[r]     = W3[r + m8];
        w3r[8 + r] = W3[16 + r + m8];
    }
    const float b3s = b3[0];

    // deriv: k = W3^T tanh(W2^T tanh(base + w1y*y) + b2) + b3
    auto deriv = [&](const v8f& ba, const v8f& bb, float yc) -> float {
        // stage 1 tail: rank-1 y update + tanh (permuted D-layout regs)
        float t[16];
        #pragma unroll
        for (int r = 0; r < 8; ++r) t[r]     = tanh_hw(fmaf(w1y[r],     yc, ba[r]));
        #pragma unroll
        for (int r = 0; r < 8; ++r) t[8 + r] = tanh_hw(fmaf(w1y[8 + r], yc, bb[r]));

        // pack to f16 pairs: thanks to the unit relabeling this IS the
        // stage-2 B matrix -- no permlane, no cndmask.
        v8u bu;
        #pragma unroll
        for (int j = 0; j < 4; ++j) {
            bu[j]     = pk16(t[2 * j],     t[2 * j + 1]);
            bu[4 + j] = pk16(t[8 + 2 * j], t[9 + 2 * j]);
        }
        v16h bmat = __builtin_bit_cast(v16h, bu);

        // stage 2: one WMMA per 16-row half of W2^T, bias via C operand
        v8f d2a = __builtin_amdgcn_wmma_f32_16x16x32_f16(
            false, a2a, false, bmat, (short)0, cb2a, false, false);
        v8f d2b = __builtin_amdgcn_wmma_f32_16x16x32_f16(
            false, a2b, false, bmat, (short)0, cb2b, false, false);

        // stage 3: 32->1 projection. 4 independent FMA accumulators (depth 4)
        // + 2-level add tree instead of a serial 16-deep chain.
        float p0 = 0.f, p1 = 0.f, p2 = 0.f, p3 = 0.f;
        #pragma unroll
        for (int r = 0; r < 2; ++r) {
            p0 = fmaf(w3r[r],          tanh_hw(d2a[r]),          p0);
            p1 = fmaf(w3r[2 + r],      tanh_hw(d2a[2 + r]),      p1);
            p2 = fmaf(w3r[4 + r],      tanh_hw(d2a[4 + r]),      p2);
            p3 = fmaf(w3r[6 + r],      tanh_hw(d2a[6 + r]),      p3);
            p0 = fmaf(w3r[8 + r],      tanh_hw(d2b[r]),          p0);
            p1 = fmaf(w3r[10 + r],     tanh_hw(d2b[2 + r]),      p1);
            p2 = fmaf(w3r[12 + r],     tanh_hw(d2b[4 + r]),      p2);
            p3 = fmaf(w3r[14 + r],     tanh_hw(d2b[6 + r]),      p3);
        }
        float acc = (p0 + p1) + (p2 + p3);
        acc += xhalf_f(acc);            // combine the two hid-out halves
        return acc + b3s;
    };

    float  y  = 0.f;
    float2 xc = *(const float2*)(x + (size_t)(b0 + M) * F_CT);   // s = 0

    #pragma unroll 1
    for (int s = 0; s < S_LEN; ++s) {
        // software prefetch next step's features
        const int sn = (s + 1 < S_LEN) ? (s + 1) : s;
        float2 xn = *(const float2*)(x + ((size_t)sn * B_SZ + b0 + M) * F_CT);

        // RK4-stage-invariant base = W1^T[:,0:2] @ x^T + b1  (f32 WMMA, K=4)
        v2f bx;
        bx[0] = lo ? xc.x : 0.f;   // K=0 row (hi lanes carry zero pad K=2,3)
        bx[1] = lo ? xc.y : 0.f;   // K=1 row
        v8f ba = __builtin_amdgcn_wmma_f32_16x16x4_f32(
            false, a1a, false, bx, (short)0, cb1a, false, false);
        v8f bb = __builtin_amdgcn_wmma_f32_16x16x4_f32(
            false, a1b, false, bx, (short)0, cb1b, false, false);

        const float k1 = deriv(ba, bb, y);
        const float k2 = deriv(ba, bb, fmaf(0.5f, k1, y));
        const float k3 = deriv(ba, bb, fmaf(0.5f, k2, y));
        const float k4 = deriv(ba, bb, y + k3);
        y = fmaf(k1 + 2.f * (k2 + k3) + k4, 1.f / 6.f, y);

        if (lo) out[(size_t)s * B_SZ + b0 + M] = y;
        xc = xn;
    }
}

extern "C" void kernel_launch(void* const* d_in, const int* in_sizes, int n_in,
                              void* d_out, int out_size, void* d_ws, size_t ws_size,
                              hipStream_t stream) {
    const float* x  = (const float*)d_in[0];
    const float* W1 = (const float*)d_in[1];
    const float* b1 = (const float*)d_in[2];
    const float* W2 = (const float*)d_in[3];
    const float* b2 = (const float*)d_in[4];
    const float* W3 = (const float*)d_in[5];
    const float* b3 = (const float*)d_in[6];
    float* out = (float*)d_out;

    dim3 grid(B_SZ / 16);   // 64 independent 16-batch ODE tiles -> 64 waves
    dim3 block(32);         // one wave32 per workgroup, spread across WGPs
    rk4_ode_wmma_kernel<<<grid, block, 0, stream>>>(x, W1, b1, W2, b2, W3, b3, out);
}
